// PostHocAffineDualSoftMaxMatcher_61014305407063
// MI455X (gfx1250) — compile-verified, matrix operand order checked by reading, not run on hardware
//
#include <hip/hip_runtime.h>
#include <hip/hip_bf16.h>

// ---------------------------------------------------------------------------
// PostHocAffineDualSoftMaxMatcher for MI455X (gfx1250, wave32, WMMA)
// B=4, N=M=4096, D=256, D0=128, K=64
// ---------------------------------------------------------------------------

typedef __attribute__((ext_vector_type(16))) __bf16        v16bf;
typedef __attribute__((ext_vector_type(8)))  float         v8f;
typedef __attribute__((ext_vector_type(4)))  unsigned int  v4u;

#define INV_TEMP   20.0f
#define THRESHOLD  0.01f
#define SING_CUT   3.0f
#define EPS_G      1e-6f

// padded LDS row stride (elements): 256 data + 8 pad = 528B -> rows land on
// distinct bank groups (4*row mod 64 distinct for row=0..15)
#define BT_STRIDE 264

// Use inline-asm CDNA5 async global->LDS copies (ASYNCcnt-tracked).
#define CDNA5_ASYNC_ASM 1

__device__ __forceinline__ unsigned short f2bf(float x) {
  unsigned int u = __float_as_uint(x);
  unsigned int r = (u + 0x7FFFu + ((u >> 16) & 1u)) >> 16;
  return (unsigned short)r;
}

union FragBF { v16bf v; v4u q[2]; };

// Global fragment load per ISA 7.12.2 16-bit A/B layout.
__device__ __forceinline__ v16bf load_frag(const unsigned short* __restrict__ row,
                                           int koff, int hk) {
  FragBF f;
  f.q[0] = *(const v4u*)(row + koff + hk * 8);
  f.q[1] = *(const v4u*)(row + koff + 16 + hk * 8);
  return f.v;
}

// LDS fragment load (row-padded tile) into a FragBF.
__device__ __forceinline__ void load_frag_lds(FragBF& f, const unsigned short* base,
                                              int row, int koff, int hk) {
  const unsigned short* p = base + row * BT_STRIDE + koff + hk * 8;
  f.q[0] = *(const v4u*)p;
  f.q[1] = *(const v4u*)(p + 16);
}

// Load all 8 B fragments of a tile and pin them live with an empty-asm
// dependency fence so the 16 ds_load_b128s issue as a clause and drain with
// partial dscnt waits underneath the WMMA chain (instead of wait-0 each).
__device__ __forceinline__ void load_btile_frags(FragBF (&fb)[8],
                                                 const unsigned short* lb,
                                                 int m16, int hk) {
#pragma unroll
  for (int kb = 0; kb < 8; ++kb) load_frag_lds(fb[kb], lb, m16, kb * 32, hk);
  asm volatile(""
               : "+v"(fb[0].q[0]), "+v"(fb[0].q[1]),
                 "+v"(fb[1].q[0]), "+v"(fb[1].q[1]),
                 "+v"(fb[2].q[0]), "+v"(fb[2].q[1]),
                 "+v"(fb[3].q[0]), "+v"(fb[3].q[1]),
                 "+v"(fb[4].q[0]), "+v"(fb[4].q[1]),
                 "+v"(fb[5].q[0]), "+v"(fb[5].q[1]),
                 "+v"(fb[6].q[0]), "+v"(fb[6].q[1]),
                 "+v"(fb[7].q[0]), "+v"(fb[7].q[1]));
}

// 16B global -> LDS copy. Async (GLOBAL_LOAD_ASYNC_TO_LDS_B128, ASYNCcnt) via
// inline asm; low 32 bits of a generic __shared__ pointer are the LDS byte
// offset (ISA 10.2 aperture rules). Fallback: VGPR relay.
__device__ __forceinline__ void g2l16(const unsigned short* g, unsigned short* l) {
#if CDNA5_ASYNC_ASM
  asm volatile("global_load_async_to_lds_b128 %0, %1, off"
               :
               : "v"((unsigned int)(size_t)l),
                 "v"((unsigned long long)(size_t)g)
               : "memory");
#else
  *(v4u*)l = *(const v4u*)g;
#endif
}

__device__ __forceinline__ void async_fence() {
#if CDNA5_ASYNC_ASM
#if defined(__has_builtin) && __has_builtin(__builtin_amdgcn_s_wait_asynccnt)
  __builtin_amdgcn_s_wait_asynccnt(0);
#else
  asm volatile("s_wait_asynccnt 0x0" ::: "memory");
#endif
#endif
}

// Stage one 16x256 bf16 B tile into LDS cooperatively (256 threads x 2 chunks).
__device__ __forceinline__ void stage_tile(const unsigned short* __restrict__ Bbase,
                                           int mt, unsigned short* ldst, int tid) {
#pragma unroll
  for (int c = 0; c < 2; ++c) {
    int ch = tid + c * 256;                 // 0..511
    int r  = ch >> 5;                       // tile row 0..15
    int o8 = (ch & 31) * 8;                 // 16B chunk offset in elements
    const unsigned short* g = Bbase + (((size_t)(mt * 16 + r)) << 8) + o8;
    g2l16(g, ldst + r * BT_STRIDE + o8);
  }
}

// ---------------------------------------------------------------------------
// K0: zero the column-max-bits buffer
// ---------------------------------------------------------------------------
__global__ void zero_i32_kernel(int* __restrict__ p, int n) {
  int i = blockIdx.x * 256 + threadIdx.x;
  if (i < n) p[i] = 0;
}

// ---------------------------------------------------------------------------
// K1: L2-normalize each 256-dim descriptor row -> bf16; also emit raw first
// 128 dims (desc0) as bf16 for the softplus projection GEMM.
// ---------------------------------------------------------------------------
__global__ void __launch_bounds__(256) normalize_kernel(
    const float* __restrict__ desc, unsigned short* __restrict__ outbf,
    unsigned short* __restrict__ out0bf, int totalRows) {
  const int lane = threadIdx.x & 31;
  const int wave = threadIdx.x >> 5;
  const int row  = blockIdx.x * 8 + wave;
  if (row >= totalRows) return;
  const float* r = desc + (size_t)row * 256;
  float x[8];
  float ss = 0.0f;
#pragma unroll
  for (int i = 0; i < 8; ++i) { x[i] = r[lane + 32 * i]; ss += x[i] * x[i]; }
#pragma unroll
  for (int off = 16; off >= 1; off >>= 1) ss += __shfl_xor(ss, off, 32);
  const float inv = 1.0f / sqrtf(ss);
  unsigned short* o = outbf + (size_t)row * 256;
#pragma unroll
  for (int i = 0; i < 8; ++i) o[lane + 32 * i] = f2bf(x[i] * inv);
  unsigned short* o0 = out0bf + (size_t)row * 128;
#pragma unroll
  for (int i = 0; i < 4; ++i) o0[lane + 32 * i] = f2bf(x[i]);  // raw desc0
}

// ---------------------------------------------------------------------------
// K1c: transpose + bf16-convert W_lstsq (128x64) -> Wt (64x128, K-contiguous)
// ---------------------------------------------------------------------------
__global__ void wtranspose_kernel(const float* __restrict__ W,
                                  unsigned short* __restrict__ Wt) {
  int i = blockIdx.x * 256 + threadIdx.x;
  if (i < 128 * 64) {
    int k = i >> 6, c = i & 63;
    Wt[(size_t)c * 128 + k] = f2bf(W[i]);
  }
}

// ---------------------------------------------------------------------------
// K2: streaming WMMA pass computing ONLINE softmax stats per row.
// Block = 8 waves = 128 rows of one batch; B tiles double-buffered in LDS
// via async global->LDS; all 8 B fragments pinned live before the WMMA
// chain (two accumulator chains for matrix-pipe ILP).
// Launched twice: (dA,dB) -> row stats; (dB,dA) -> column stats.
// ---------------------------------------------------------------------------
__global__ void __launch_bounds__(256) sim_stats_kernel(
    const unsigned short* __restrict__ Abf, const unsigned short* __restrict__ Bbf,
    float* __restrict__ omax, float* __restrict__ osum, int rowsN, int colsM) {
  __shared__ unsigned short Btile[2][16 * BT_STRIDE];
  const int tid  = threadIdx.x;
  const int lane = tid & 31;
  const int wave = tid >> 5;
  const int blocksPerBatch = rowsN >> 7;
  const int b    = blockIdx.x / blocksPerBatch;
  const int rblk = blockIdx.x % blocksPerBatch;
  const int row0 = rblk * 128 + wave * 16;
  const int m16 = lane & 15;
  const int hk  = lane >> 4;

  const unsigned short* Arow = Abf + (((size_t)b * rowsN + row0 + m16) << 8);
  v16bf afrag[8];
#pragma unroll
  for (int kb = 0; kb < 8; ++kb) afrag[kb] = load_frag(Arow, kb * 32, hk);

  float run_max[8], run_sum[8];
#pragma unroll
  for (int i = 0; i < 8; ++i) { run_max[i] = -3.0e38f; run_sum[i] = 0.0f; }

  const unsigned short* Bbase = Bbf + (((size_t)b * colsM) << 8);
  const int mtiles = colsM >> 4;

  stage_tile(Bbase, 0, &Btile[0][0], tid);
  async_fence();
  __syncthreads();

  for (int mt = 0; mt < mtiles; ++mt) {
    const int cur = mt & 1;
    if (mt + 1 < mtiles) stage_tile(Bbase, mt + 1, &Btile[cur ^ 1][0], tid);

    FragBF fb[8];
    load_btile_frags(fb, &Btile[cur][0], m16, hk);

    v8f acc0 = {}, acc1 = {};
#pragma unroll
    for (int kb = 0; kb < 8; kb += 2) {
      acc0 = __builtin_amdgcn_wmma_f32_16x16x32_bf16(
          false, afrag[kb], false, fb[kb].v, (short)0, acc0, false, false);
      acc1 = __builtin_amdgcn_wmma_f32_16x16x32_bf16(
          false, afrag[kb + 1], false, fb[kb + 1].v, (short)0, acc1, false, false);
    }
#pragma unroll
    for (int i = 0; i < 8; ++i) {
      float v  = INV_TEMP * (acc0[i] + acc1[i]);
      float nm = fmaxf(run_max[i], v);
      run_sum[i] = run_sum[i] * __expf(run_max[i] - nm) + __expf(v - nm);
      run_max[i] = nm;
    }
    async_fence();
    __syncthreads();
  }
  // combine the 16 column-lanes (max/sum pairs) within each half-wave
#pragma unroll
  for (int off = 1; off < 16; off <<= 1) {
#pragma unroll
    for (int i = 0; i < 8; ++i) {
      float om = __shfl_xor(run_max[i], off, 32);
      float os = __shfl_xor(run_sum[i], off, 32);
      float nm = fmaxf(run_max[i], om);
      run_sum[i] = run_sum[i] * __expf(run_max[i] - nm) + os * __expf(om - nm);
      run_max[i] = nm;
    }
  }
  if (m16 == 0) {
    size_t o = (size_t)b * rowsN + row0 + hk * 8;
#pragma unroll
    for (int i = 0; i < 8; ++i) { omax[o + i] = run_max[i]; osum[o + i] = run_sum[i]; }
  }
}

// ---------------------------------------------------------------------------
// K3: recompute sim tiles (same LDS-staged pipeline), form
// P = exp(2s - rmax - cmax)/(rsum*csum), per-row argmax in registers,
// per-column max via LDS reduction + one integer atomicMax per column.
// ---------------------------------------------------------------------------
__global__ void __launch_bounds__(256) ppass_kernel(
    const unsigned short* __restrict__ Abf, const unsigned short* __restrict__ Bbf,
    const float* __restrict__ rmax, const float* __restrict__ rsum,
    const float* __restrict__ cmax, const float* __restrict__ csum,
    int* __restrict__ colmaxbits, int* __restrict__ jout,
    float* __restrict__ pijout, int N, int M) {
  __shared__ unsigned short Btile[2][16 * BT_STRIDE];
  __shared__ float smax[8][16];
  const int tid  = threadIdx.x;
  const int lane = tid & 31;
  const int wave = tid >> 5;
  const int blocksPerBatch = N >> 7;
  const int b    = blockIdx.x / blocksPerBatch;
  const int rblk = blockIdx.x % blocksPerBatch;
  const int row0 = rblk * 128 + wave * 16;
  const int m16 = lane & 15;
  const int hk  = lane >> 4;

  const unsigned short* Arow = Abf + (((size_t)b * N + row0 + m16) << 8);
  v16bf afrag[8];
#pragma unroll
  for (int kb = 0; kb < 8; ++kb) afrag[kb] = load_frag(Arow, kb * 32, hk);

  const int rbase = row0 + hk * 8;
  float rm[8], rinv[8];
#pragma unroll
  for (int i = 0; i < 8; ++i) {
    rm[i]   = rmax[(size_t)b * N + rbase + i];
    rinv[i] = 1.0f / rsum[(size_t)b * N + rbase + i];
  }

  float best_p[8];
  int   best_j[8];
#pragma unroll
  for (int i = 0; i < 8; ++i) { best_p[i] = -1.0f; best_j[i] = 0; }

  const unsigned short* Bbase = Bbf + (((size_t)b * M) << 8);
  const int mtiles = M >> 4;

  stage_tile(Bbase, 0, &Btile[0][0], tid);
  async_fence();
  __syncthreads();

  for (int mt = 0; mt < mtiles; ++mt) {
    const int cur = mt & 1;
    if (mt + 1 < mtiles) stage_tile(Bbase, mt + 1, &Btile[cur ^ 1][0], tid);

    FragBF fb[8];
    load_btile_frags(fb, &Btile[cur][0], m16, hk);

    v8f acc0 = {}, acc1 = {};
#pragma unroll
    for (int kb = 0; kb < 8; kb += 2) {
      acc0 = __builtin_amdgcn_wmma_f32_16x16x32_bf16(
          false, afrag[kb], false, fb[kb].v, (short)0, acc0, false, false);
      acc1 = __builtin_amdgcn_wmma_f32_16x16x32_bf16(
          false, afrag[kb + 1], false, fb[kb + 1].v, (short)0, acc1, false, false);
    }
    const int col = mt * 16 + m16;
    const float cm   = cmax[(size_t)b * M + col];
    const float cinv = 1.0f / csum[(size_t)b * M + col];
    float tmax = 0.0f;
#pragma unroll
    for (int i = 0; i < 8; ++i) {
      float s = INV_TEMP * (acc0[i] + acc1[i]);
      float p = __expf(2.0f * s - rm[i] - cm) * rinv[i] * cinv;
      if (p > best_p[i]) { best_p[i] = p; best_j[i] = col; }
      tmax = fmaxf(tmax, p);
    }
    // column max across the two 8-row halves of this wave's tile
    tmax = fmaxf(tmax, __shfl_xor(tmax, 16, 32));
    if (lane < 16) smax[wave][lane] = tmax;
    async_fence();
    __syncthreads();
    if (tid < 16) {
      float mval = smax[0][tid];
#pragma unroll
      for (int w = 1; w < 8; ++w) mval = fmaxf(mval, smax[w][tid]);
      atomicMax(&colmaxbits[(size_t)b * M + mt * 16 + tid],
                __float_as_int(mval));  // P >= 0 -> bit order == float order
    }
    __syncthreads();
  }
  // per-row argmax across the 16 column-lanes (first-index tie break)
#pragma unroll
  for (int off = 1; off < 16; off <<= 1) {
#pragma unroll
    for (int i = 0; i < 8; ++i) {
      float op = __shfl_xor(best_p[i], off, 32);
      int   oj = __shfl_xor(best_j[i], off, 32);
      if (op > best_p[i] || (op == best_p[i] && oj < best_j[i])) {
        best_p[i] = op; best_j[i] = oj;
      }
    }
  }
  if (m16 == 0) {
    size_t o = (size_t)b * N + rbase;
#pragma unroll
    for (int i = 0; i < 8; ++i) { jout[o + i] = best_j[i]; pijout[o + i] = best_p[i]; }
  }
}

// ---------------------------------------------------------------------------
// K4: softplus(desc0 @ W + b) for all rows, via WMMA (K=128, 4 steps).
// ---------------------------------------------------------------------------
__global__ void __launch_bounds__(256) wgemm_kernel(
    const unsigned short* __restrict__ Xbf /* rows x 128 */,
    const unsigned short* __restrict__ Wtbf /* 64 x 128 */,
    const float* __restrict__ bvec, float* __restrict__ outw /* rows x 64 */,
    int rows) {
  const int lane = threadIdx.x & 31;
  const int wave = threadIdx.x >> 5;
  const int wid  = blockIdx.x * 8 + wave;
  const int rt = wid >> 2;
  const int ct = wid & 3;
  if (rt * 16 >= rows) return;
  const int m16 = lane & 15;
  const int hk  = lane >> 4;
  const unsigned short* Xrow = Xbf + (size_t)(rt * 16 + m16) * 128;
  const unsigned short* Wrow = Wtbf + (size_t)(ct * 16 + m16) * 128;
  v16bf a0 = load_frag(Xrow, 0, hk);
  v16bf w0 = load_frag(Wrow, 0, hk);
  v16bf a1 = load_frag(Xrow, 32, hk);
  v16bf w1 = load_frag(Wrow, 32, hk);
  v16bf a2 = load_frag(Xrow, 64, hk);
  v16bf w2 = load_frag(Wrow, 64, hk);
  v16bf a3 = load_frag(Xrow, 96, hk);
  v16bf w3 = load_frag(Wrow, 96, hk);
  v8f acc0 = {}, acc1 = {};
  acc0 = __builtin_amdgcn_wmma_f32_16x16x32_bf16(false, a0, false, w0, (short)0, acc0, false, false);
  acc1 = __builtin_amdgcn_wmma_f32_16x16x32_bf16(false, a1, false, w1, (short)0, acc1, false, false);
  acc0 = __builtin_amdgcn_wmma_f32_16x16x32_bf16(false, a2, false, w2, (short)0, acc0, false, false);
  acc1 = __builtin_amdgcn_wmma_f32_16x16x32_bf16(false, a3, false, w3, (short)0, acc1, false, false);
  const int col = ct * 16 + m16;
  const float bb = bvec[col];
  const int rbase = rt * 16 + hk * 8;
#pragma unroll
  for (int i = 0; i < 8; ++i) {
    float x  = acc0[i] + acc1[i] + bb;
    float sp = (x > 20.0f) ? x : __logf(1.0f + __expf(x));  // softplus
    outw[(size_t)(rbase + i) * 64 + col] = sp;
  }
}

// ---------------------------------------------------------------------------
// K5: per-match finalize: mutual-NN check, weighted 2x2 LSQ affine solve,
// closed-form 2x2 SVD gating, outputs.
// ---------------------------------------------------------------------------
__global__ void __launch_bounds__(256) finalize_kernel(
    const float* __restrict__ kpA, const float* __restrict__ kpB,
    const float* __restrict__ descA, const float* __restrict__ descB,
    const int* __restrict__ jbuf, const float* __restrict__ pij,
    const int* __restrict__ colmaxbits,
    const float* __restrict__ wA, const float* __restrict__ wB,
    float* __restrict__ mA, float* __restrict__ mB,
    float* __restrict__ jout_f, float* __restrict__ valid_f, int N, int M) {
  const size_t idx = (size_t)blockIdx.x * 256 + threadIdx.x;
  const int b = (int)(idx / N);
  const int j = jbuf[idx];
  const float P   = pij[idx];
  const float cmj = __int_as_float(colmaxbits[(size_t)b * M + j]);
  bool valid = (P >= cmj) && (P > THRESHOLD);

  const float* x1 = descA + idx * 256 + 128;
  const float* y1 = descB + ((size_t)b * M + j) * 256 + 128;
  const float* wa = wA + idx * 64;
  const float* wb = wB + ((size_t)b * M + j) * 64;

  float G00 = 0, G01 = 0, G11 = 0, C00 = 0, C01 = 0, C10 = 0, C11 = 0;
#pragma unroll 4
  for (int k = 0; k < 64; ++k) {
    float w  = sqrtf(wa[k] * wb[k]);
    float x0 = x1[2 * k], xx1 = x1[2 * k + 1];
    float y0 = y1[2 * k], yy1 = y1[2 * k + 1];
    G00 += w * x0 * x0;  G01 += w * x0 * xx1;  G11 += w * xx1 * xx1;
    C00 += w * y0 * x0;  C01 += w * y0 * xx1;
    C10 += w * yy1 * x0; C11 += w * yy1 * xx1;
  }
  G00 += EPS_G; G11 += EPS_G;
  float idet = 1.0f / (G00 * G11 - G01 * G01);
  // est = C * G^{-1}  (G symmetric)
  float e00 = ( C00 * G11 - C01 * G01) * idet;
  float e01 = (-C00 * G01 + C01 * G00) * idet;
  float e10 = ( C10 * G11 - C11 * G01) * idet;
  float e11 = (-C10 * G01 + C11 * G00) * idet;
  // singular values of 2x2 via est^T est
  float M00 = e00 * e00 + e10 * e10;
  float M01 = e00 * e01 + e10 * e11;
  float M11 = e01 * e01 + e11 * e11;
  float t = M00 + M11;
  float d = M00 * M11 - M01 * M01;
  float disc = sqrtf(fmaxf(0.25f * t * t - d, 0.0f));
  float shi = sqrtf(0.5f * t + disc);
  float slo = sqrtf(fmaxf(0.5f * t - disc, 0.0f));
  float det = e00 * e11 - e01 * e10;
  float sumsv = shi + slo;
  bool good = isfinite(sumsv) && (shi < SING_CUT) && (slo > 1.0f / SING_CUT) && (det > 0.0f);
  valid = valid && good;

  mA[idx * 2]     = kpA[idx * 2];
  mA[idx * 2 + 1] = kpA[idx * 2 + 1];
  const float* kb2 = kpB + ((size_t)b * M + j) * 2;
  mB[idx * 2]     = kb2[0];
  mB[idx * 2 + 1] = kb2[1];
  jout_f[idx]  = (float)j;
  valid_f[idx] = valid ? 1.0f : 0.0f;
}

// ---------------------------------------------------------------------------
// Host launcher
// ---------------------------------------------------------------------------
extern "C" void kernel_launch(void* const* d_in, const int* in_sizes, int n_in,
                              void* d_out, int out_size, void* d_ws, size_t ws_size,
                              hipStream_t stream) {
  (void)in_sizes; (void)n_in; (void)out_size; (void)ws_size;
  const int B = 4, N = 4096, M = 4096;

  const float* kpA   = (const float*)d_in[0];
  const float* descA = (const float*)d_in[1];
  const float* kpB   = (const float*)d_in[2];
  const float* descB = (const float*)d_in[3];
  const float* Wl    = (const float*)d_in[4];
  const float* bl    = (const float*)d_in[5];

  char* ws = (char*)d_ws;
  size_t off = 0;
  auto take = [&](size_t bytes) -> char* {
    char* p = ws + off;
    off = (off + bytes + 255) & ~(size_t)255;
    return p;
  };
  unsigned short* dAbf  = (unsigned short*)take((size_t)B * N * 256 * 2);
  unsigned short* dBbf  = (unsigned short*)take((size_t)B * M * 256 * 2);
  unsigned short* d0Abf = (unsigned short*)take((size_t)B * N * 128 * 2);
  unsigned short* d0Bbf = (unsigned short*)take((size_t)B * M * 128 * 2);
  unsigned short* Wtbf  = (unsigned short*)take((size_t)128 * 64 * 2);
  float* rmaxb  = (float*)take((size_t)B * N * 4);
  float* rsumb  = (float*)take((size_t)B * N * 4);
  float* cmaxb  = (float*)take((size_t)B * M * 4);
  float* csumb  = (float*)take((size_t)B * M * 4);
  int*   cmbits = (int*)  take((size_t)B * M * 4);
  int*   jbuf   = (int*)  take((size_t)B * N * 4);
  float* pijb   = (float*)take((size_t)B * N * 4);
  float* wAb    = (float*)take((size_t)B * N * 64 * 4);
  float* wBb    = (float*)take((size_t)B * M * 64 * 4);

  float* out = (float*)d_out;
  float* mA = out;
  float* mB = out + (size_t)B * N * 2;
  float* jf = out + (size_t)B * N * 4;
  float* vf = out + (size_t)B * N * 5;

  // K0: init column-max bits (P >= 0 so bits(0.0f)==0 works)
  zero_i32_kernel<<<(B * M + 255) / 256, 256, 0, stream>>>(cmbits, B * M);

  // K1: normalize + bf16 convert (and raw desc0 bf16)
  normalize_kernel<<<(B * N) / 8, 256, 0, stream>>>(descA, dAbf, d0Abf, B * N);
  normalize_kernel<<<(B * M) / 8, 256, 0, stream>>>(descB, dBbf, d0Bbf, B * M);
  wtranspose_kernel<<<(128 * 64 + 255) / 256, 256, 0, stream>>>(Wl, Wtbf);

  // K2: row stats of sim, then of sim^T (= column stats of sim)
  sim_stats_kernel<<<B * (N / 128), 256, 0, stream>>>(dAbf, dBbf, rmaxb, rsumb, N, M);
  sim_stats_kernel<<<B * (M / 128), 256, 0, stream>>>(dBbf, dAbf, cmaxb, csumb, M, N);

  // K4: softplus projection weights for all A rows and all B rows
  wgemm_kernel<<<(B * N) / 32, 256, 0, stream>>>(d0Abf, Wtbf, bl, wAb, B * N);
  wgemm_kernel<<<(B * M) / 32, 256, 0, stream>>>(d0Bbf, Wtbf, bl, wBb, B * M);

  // K3: P pass -> argmax j, Pij, column maxima of P
  ppass_kernel<<<B * (N / 128), 256, 0, stream>>>(dAbf, dBbf, rmaxb, rsumb,
                                                  cmaxb, csumb, cmbits, jbuf,
                                                  pijb, N, M);

  // K5: finalize matches + affine gating
  finalize_kernel<<<(B * N) / 256, 256, 0, stream>>>(kpA, kpB, descA, descB,
                                                     jbuf, pijb, cmbits, wAb, wBb,
                                                     mA, mB, jf, vf, N, M);
}
